// DivFreeNetwork_14379550507520
// MI455X (gfx1250) — matrix operand order repaired and checked by reading, not compiled, not run on hardware
//
#include <hip/hip_runtime.h>
#include <hip/hip_bf16.h>

typedef __attribute__((ext_vector_type(16))) _Float16 v16h;
typedef __attribute__((ext_vector_type(8)))  _Float16 v8h;
typedef __attribute__((ext_vector_type(8)))  float    v8f;

#define D_DIM     256
#define SAMP_WG   8
#define ROWS      32           // 8 samples * 4 channels (value + 3 tangents)
#define H_STRIDE  264          // halves per row (+8 pad -> conflict-free ds_read_b128)
#define P_STRIDE  260          // floats per row (+4 pad)

// ---------------------------------------------------------------------------
// One-time: convert fp32 weights [mat][k][n] -> f16 transposed [mat][n][k]
// so WMMA B-fragments are contiguous 32B loads per lane (L2-resident).
// ---------------------------------------------------------------------------
__global__ void transpose_to_f16(const float* __restrict__ w,
                                 _Float16* __restrict__ wT, int total) {
    int idx = blockIdx.x * 256 + threadIdx.x;
    if (idx >= total) return;
    int mat = idx >> 16;
    int n   = (idx >> 8) & 255;
    int k   = idx & 255;
    wT[(mat << 16) + (n << 8) + k] = (_Float16)w[(mat << 16) + (k << 8) + n];
}

// ---------------------------------------------------------------------------
// Fused forward + 3 forward-mode tangent passes through the MLP, then the
// softmax-mixture Jacobian reduction -> divergence-free field u (3 per sample).
// ---------------------------------------------------------------------------
__launch_bounds__(128)
__global__ void divfree_mlp(const float* __restrict__ x,
                            const float* __restrict__ w_in,
                            const float* __restrict__ b_in,
                            const float* __restrict__ b_hid,
                            const float* __restrict__ b_out,
                            const _Float16* __restrict__ whidT,
                            const _Float16* __restrict__ woutT,
                            float* __restrict__ out) {
    __shared__ _Float16 lds_h[ROWS * H_STRIDE];   // f16 activation panel (WMMA A)
    __shared__ float    lds_pre[ROWS * P_STRIDE]; // f32 pre-activation panel

    const int tid     = threadIdx.x;
    const int lane    = tid & 31;
    const int wave    = tid >> 5;
    const int l16     = lane & 15;
    const int hi      = lane >> 4;            // lane half (0: lanes 0-15, 1: 16-31)
    const int sample0 = blockIdx.x * SAMP_WG;

    // ---- layer 0 fused with its activation (K=3, plain VALU) ---------------
    // value row: silu(z @ w_in + b_in); tangent rows: w_in[j] * silu'(pre)
    for (int i = tid; i < SAMP_WG * D_DIM; i += 128) {
        int s = i >> 8, col = i & 255;
        float w0 = w_in[col], w1 = w_in[256 + col], w2 = w_in[512 + col];
        const float* z = x + (size_t)(sample0 + s) * 3;
        float xv  = z[0] * w0 + z[1] * w1 + z[2] * w2 + b_in[col];
        float sig = 1.0f / (1.0f + __expf(-xv));
        float dsl = sig * (1.0f + xv * (1.0f - sig));   // silu'
        _Float16* hp = lds_h + (s * 4) * H_STRIDE + col;
        hp[0]            = (_Float16)(xv * sig);        // silu
        hp[H_STRIDE]     = (_Float16)(w0 * dsl);
        hp[2 * H_STRIDE] = (_Float16)(w1 * dsl);
        hp[3 * H_STRIDE] = (_Float16)(w2 * dsl);
    }

    // ---- mid-layer activation: one sigmoid per value element, reused by the
    //      3 tangent rows (JVP chain rule: t_out = t_in * silu'(pre_value)) ---
    auto act_stage = [&](const float* __restrict__ bias) {
        __syncthreads();
        for (int i = tid; i < SAMP_WG * D_DIM; i += 128) {
            int s = i >> 8, col = i & 255;
            const float* pp = lds_pre + (s * 4) * P_STRIDE + col;
            float xv  = pp[0] + bias[col];
            float sig = 1.0f / (1.0f + __expf(-xv));
            float dsl = sig * (1.0f + xv * (1.0f - sig));
            _Float16* hp = lds_h + (s * 4) * H_STRIDE + col;
            hp[0]            = (_Float16)(xv * sig);
            hp[H_STRIDE]     = (_Float16)(pp[P_STRIDE]     * dsl);
            hp[2 * H_STRIDE] = (_Float16)(pp[2 * P_STRIDE] * dsl);
            hp[3 * H_STRIDE] = (_Float16)(pp[3 * P_STRIDE] * dsl);
        }
        __syncthreads();
    };

    // ---- one 256x256 layer: [32x256] @ [256x256] via v_wmma_f32_16x16x32_f16
    auto layer_wmma = [&](const _Float16* __restrict__ wT) {
        const int rowbase = (wave & 1) * 16;      // row-tile per wave
        const int nt0     = (wave >> 1) * 8;      // col-tile half per wave
        const _Float16* aptr = lds_h + (rowbase + l16) * H_STRIDE + hi * 16;
        for (int nt = nt0; nt < nt0 + 8; ++nt) {
            v8f acc = {};
            const _Float16* bptr = wT + (size_t)(nt * 16 + l16) * 256 + hi * 16;
#pragma unroll
            for (int kt = 0; kt < 8; ++kt) {
                // A 16x32 f16 fragment: halves 0-15 = K kbase+16*hi..+15 (contiguous)
                v8h alo = *(const v8h*)(aptr + kt * 32);
                v8h ahi = *(const v8h*)(aptr + kt * 32 + 8);
                v16h a;
#pragma unroll
                for (int q = 0; q < 8; ++q) { a[q] = alo[q]; a[8 + q] = ahi[q]; }
                // B 32x16 f16 fragment: 16 contiguous K halves (transposed weights)
                v16h b = *(const v16h*)(bptr + kt * 32);
                acc = __builtin_amdgcn_wmma_f32_16x16x32_f16(
                        false, a, false, b, (short)0, acc, false, false);
            }
            // D layout: vgpr r -> row rowbase + r + hi*8, col = nt*16 + l16
            int colw = nt * 16 + l16;
            int rw   = rowbase + hi * 8;
#pragma unroll
            for (int r = 0; r < 8; ++r)
                lds_pre[(rw + r) * P_STRIDE + colw] = acc[r];
        }
    };

    __syncthreads();                 // lds_h (layer-0 output) ready
    layer_wmma(whidT);               // hidden layer 0
    act_stage(b_hid);
    layer_wmma(whidT + 65536);       // hidden layer 1
    act_stage(b_hid + 256);
    layer_wmma(whidT + 131072);      // hidden layer 2
    act_stage(b_hid + 512);
    layer_wmma(woutT);               // output head (linear, pre in lds_pre)
    __syncthreads();

    // ---- per-sample mixture Jacobian: dt[j][k] = d tri_k / d z_j ------------
    if (tid < SAMP_WG) {
        const int s = tid;
        const float* P0 = lds_pre + (s * 4 + 0) * P_STRIDE;   // o value row (pre-bias)
        const float* T0 = lds_pre + (s * 4 + 1) * P_STRIDE;   // d o / d z0
        const float* T1 = lds_pre + (s * 4 + 2) * P_STRIDE;
        const float* T2 = lds_pre + (s * 4 + 3) * P_STRIDE;

        float mx = -3.4e38f;
        for (int m = 0; m < 64; ++m)
            mx = fmaxf(mx, P0[4 * m] + b_out[4 * m]);

        float Z = 0.f;
        float wS[3] = {0.f, 0.f, 0.f}, wTk[3] = {0.f, 0.f, 0.f};
        float A1[3][3] = {{0.f}}, A2[3][3] = {{0.f}};
        for (int m = 0; m < 64; ++m) {
            float sm = P0[4 * m] + b_out[4 * m];
            float e  = __expf(sm - mx);
            Z += e;
            float ds0 = T0[4 * m], ds1 = T1[4 * m], ds2 = T2[4 * m];
            wS[0] += e * ds0; wS[1] += e * ds1; wS[2] += e * ds2;
#pragma unroll
            for (int k = 0; k < 3; ++k) {
                float ok = P0[4 * m + 1 + k] + b_out[4 * m + 1 + k];
                wTk[k] += e * ok;
                A1[0][k] += e * ds0 * ok;  A2[0][k] += e * T0[4 * m + 1 + k];
                A1[1][k] += e * ds1 * ok;  A2[1][k] += e * T1[4 * m + 1 + k];
                A1[2][k] += e * ds2 * ok;  A2[2][k] += e * T2[4 * m + 1 + k];
            }
        }
        float inv = 1.0f / Z;
        float dt[3][3];
#pragma unroll
        for (int j = 0; j < 3; ++j)
#pragma unroll
            for (int k = 0; k < 3; ++k)
                dt[j][k] = inv * (A1[j][k] + A2[j][k])
                         - (wS[j] * inv) * (wTk[k] * inv);

        // u_i = sum_j dA[i,j]/dz_j for antisymmetric A built from tri
        const int g = sample0 + s;
        out[g * 3 + 0] =  dt[1][0] + dt[2][1];
        out[g * 3 + 1] = -dt[0][0] + dt[2][2];
        out[g * 3 + 2] = -dt[0][1] - dt[1][2];
    }
}

extern "C" void kernel_launch(void* const* d_in, const int* in_sizes, int n_in,
                              void* d_out, int out_size, void* d_ws, size_t ws_size,
                              hipStream_t stream) {
    const float* x     = (const float*)d_in[0];
    const float* w_in  = (const float*)d_in[1];
    const float* b_in  = (const float*)d_in[2];
    const float* w_hid = (const float*)d_in[3];
    const float* b_hid = (const float*)d_in[4];
    const float* w_out = (const float*)d_in[5];
    const float* b_out = (const float*)d_in[6];

    const int n_hid = in_sizes[3];   // 3*256*256
    const int n_out = in_sizes[5];   // 256*256
    _Float16* whidT = (_Float16*)d_ws;
    _Float16* woutT = whidT + n_hid;

    transpose_to_f16<<<(n_hid + 255) / 256, 256, 0, stream>>>(w_hid, whidT, n_hid);
    transpose_to_f16<<<(n_out + 255) / 256, 256, 0, stream>>>(w_out, woutT, n_out);

    const int N = in_sizes[0] / 3;   // 65536 samples
    divfree_mlp<<<N / SAMP_WG, 128, 0, stream>>>(
        x, w_in, b_in, b_hid, b_out, whidT, woutT, (float*)d_out);
}
